// EGNN_AE_50654844289862
// MI455X (gfx1250) — matrix operand order repaired and checked by reading, not compiled, not run on hardware
//
#include <hip/hip_runtime.h>

// ---------------------------------------------------------------------------
// EGNN autoencoder forward, fused for MI455X (gfx1250, wave32, WMMA bf16).
//   edge MLP (26->128->128) on v_wmma_f32_16x16x32_bf16, fused scatter-add
//   node MLP (139->128->128) on WMMA, 128->4 head via shfl reductions.
// Layer-1 biases folded into the K-padding as a "constant 1" input row.
// Workspace: agg f32 [100000*128] + bf16 transposed/padded weights.
// ---------------------------------------------------------------------------

typedef __attribute__((ext_vector_type(16))) __bf16 v16bf;
typedef __attribute__((ext_vector_type(8)))  float  v8f;

#define NN 100000      // nodes
#define NE 1600000     // edges
#define NF 11          // node feature dim
#define HF 128         // hidden

// Full RNE conversion: used only in the one-shot weight-prep kernel.
static __device__ __forceinline__ unsigned short f2bf(float f) {
  union { float f; unsigned u; } x; x.f = f;
  unsigned r = x.u + 0x7FFFu + ((x.u >> 16) & 1u);
  return (unsigned short)(r >> 16);
}

// Cheap round-half-up conversion for activations (1-2 VALU ops).
static __device__ __forceinline__ unsigned short f2bf_fast(float f) {
  union { float f; unsigned u; } x; x.f = f;
  return (unsigned short)((x.u + 0x8000u) >> 16);
}

union FragAB {
  v16bf v;
  unsigned short s[16];
  uint4 q[2];
};

// ---------------------------------------------------------------------------
__global__ void egnn_zero_agg(float* __restrict__ agg) {
  int i = blockIdx.x * blockDim.x + threadIdx.x;
  const int stride = gridDim.x * blockDim.x;
  for (; i < NN * HF; i += stride) agg[i] = 0.0f;
}

// Transpose weights to [N][K] (K contiguous, zero padded) in bf16.
// Layer-1 bias is appended as K-row 26 (edge) / 139 (node); the matching
// activation element is 1.0, so C can start from inline zero.
__global__ void egnn_prep_weights(const float* __restrict__ eW1,
                                  const float* __restrict__ eb1,
                                  const float* __restrict__ eW2,
                                  const float* __restrict__ nW1,
                                  const float* __restrict__ nb1,
                                  const float* __restrict__ nW2,
                                  unsigned short* __restrict__ eW1T,   // [128][32]
                                  unsigned short* __restrict__ eW2T,   // [128][128]
                                  unsigned short* __restrict__ nW1T,   // [128][160]
                                  unsigned short* __restrict__ nW2T) { // [128][128]
  const int S1 = 128 * 32, S2 = 128 * 128, S3 = 128 * 160, S4 = 128 * 128;
  int t = blockIdx.x * blockDim.x + threadIdx.x;
  if (t < S1) {
    int n = t / 32, k = t % 32;
    unsigned short v = 0;
    if (k < 26)       v = f2bf(eW1[k * 128 + n]);
    else if (k == 26) v = f2bf(eb1[n]);
    eW1T[t] = v;
  } else if (t < S1 + S2) {
    int u = t - S1; int n = u / 128, k = u % 128;
    eW2T[u] = f2bf(eW2[k * 128 + n]);
  } else if (t < S1 + S2 + S3) {
    int u = t - S1 - S2; int n = u / 160, k = u % 160;
    unsigned short v = 0;
    if (k < 139)       v = f2bf(nW1[k * 128 + n]);
    else if (k == 139) v = f2bf(nb1[n]);
    nW1T[u] = v;
  } else if (t < S1 + S2 + S3 + S4) {
    int u = t - S1 - S2 - S3; int n = u / 128, k = u % 128;
    nW2T[u] = f2bf(nW2[k * 128 + n]);
  }
}

// ---------------------------------------------------------------------------
// Edge kernel: wave processes 16 edges. 8 waves / block (32 KB LDS).
__global__ __launch_bounds__(256) void egnn_edge_kernel(
    const float* __restrict__ nf, const int* __restrict__ eidx,
    const float* __restrict__ ea,
    const unsigned short* __restrict__ eW1T,
    const unsigned short* __restrict__ eW2T, const float* __restrict__ eb2,
    float* __restrict__ agg) {
  __shared__ __align__(16) unsigned short ehs[8][16 * HF];  // per-wave 16x128 bf16
  const int wslot = threadIdx.x >> 5;
  const int base  = (blockIdx.x * 8 + wslot) * 16;
  if (base >= NE) return;                 // wave-uniform; EXEC stays all-ones
  const int lane = threadIdx.x & 31;
  const int m  = lane & 15;
  const int hi = lane >> 4;

  const int e   = base + m;
  const int row = eidx[e];
  const int col = eidx[NE + e];

  // A fragment, 16x32 bf16 (ISA layout: lo lanes K in {0-7,16-23}, hi lanes +8)
  // k==26 is the bias row (constant 1.0).
  FragAB a1;
#pragma unroll
  for (int j = 0; j < 16; ++j) {
    const int k = (j < 8) ? (hi * 8 + j) : (16 + hi * 8 + (j - 8));
    float v;
    if (k < NF)            v = nf[row * NF + k];
    else if (k < 2 * NF)   v = nf[col * NF + (k - NF)];
    else if (k < 26)       v = ea[e * 4 + (k - 22)];
    else if (k == 26)      v = 1.0f;
    else                   v = 0.0f;
    a1.s[j] = f2bf_fast(v);
  }

  // Layer 1: 8 N-tiles, K=32 (one WMMA each), C starts from zero (bias in K).
  v8f acc[8];
#pragma unroll
  for (int nt = 0; nt < 8; ++nt) {
    v8f c = {};
    FragAB b;
    const uint4* bp = (const uint4*)(eW1T + ((nt * 16 + m) * 32 + hi * 8));
    b.q[0] = bp[0]; b.q[1] = bp[2];
    acc[nt] = __builtin_amdgcn_wmma_f32_16x16x32_bf16(
        false, a1.v, false, b.v, (short)0, c, false, false);
  }

  // ReLU -> bf16 -> wave-private LDS tile (becomes A for layer 2)
#pragma unroll
  for (int nt = 0; nt < 8; ++nt)
#pragma unroll
    for (int r = 0; r < 8; ++r) {
      float x = __builtin_fmaxf(acc[nt][r], 0.0f);
      ehs[wslot][(r + hi * 8) * HF + nt * 16 + m] = f2bf_fast(x);
    }

  // Layer 2: K=128 in 4 steps of 32, 8 N-tiles; bias via C splat.
  v8f acc2[8];
#pragma unroll
  for (int nt = 0; nt < 8; ++nt) {
    const float bv = eb2[nt * 16 + m];
#pragma unroll
    for (int r = 0; r < 8; ++r) acc2[nt][r] = bv;
  }
#pragma unroll
  for (int ks = 0; ks < 4; ++ks) {
    FragAB a2;
    a2.q[0] = *(const uint4*)(&ehs[wslot][m * HF + ks * 32 + hi * 8]);
    a2.q[1] = *(const uint4*)(&ehs[wslot][m * HF + ks * 32 + 16 + hi * 8]);
#pragma unroll
    for (int nt = 0; nt < 8; ++nt) {
      FragAB b;
      const uint4* bp = (const uint4*)(eW2T + ((nt * 16 + m) * 128 + ks * 32 + hi * 8));
      b.q[0] = bp[0]; b.q[1] = bp[2];
      acc2[nt] = __builtin_amdgcn_wmma_f32_16x16x32_bf16(
          false, a2.v, false, b.v, (short)0, acc2[nt], false, false);
    }
  }

  // ReLU + fused scatter-sum (f32 atomics; agg fits in L2)
  int rows8[8];
#pragma unroll
  for (int r = 0; r < 8; ++r) rows8[r] = eidx[base + r + hi * 8];
#pragma unroll
  for (int nt = 0; nt < 8; ++nt)
#pragma unroll
    for (int r = 0; r < 8; ++r) {
      float x = __builtin_fmaxf(acc2[nt][r], 0.0f);
      unsafeAtomicAdd(&agg[(size_t)rows8[r] * HF + nt * 16 + m], x);
    }
}

// ---------------------------------------------------------------------------
// Node kernel: wave processes 16 nodes. 4 waves / block (36 KB LDS).
__global__ __launch_bounds__(128) void egnn_node_kernel(
    const float* __restrict__ nf, const float* __restrict__ agg,
    const unsigned short* __restrict__ nW1T,
    const unsigned short* __restrict__ nW2T, const float* __restrict__ nb2,
    const float* __restrict__ fW, const float* __restrict__ fb,
    float* __restrict__ out) {
  __shared__ __align__(16) unsigned short nin[4][16 * 160];
  __shared__ __align__(16) unsigned short nh [4][16 * HF];
  const int wslot = threadIdx.x >> 5;
  const int base  = (blockIdx.x * 4 + wslot) * 16;
  if (base >= NN) return;
  const int lane = threadIdx.x & 31;
  const int m  = lane & 15;
  const int hi = lane >> 4;

  // Stage n_in = concat(node_feats[11], agg[128], bias-1 at k=139), K=160 bf16.
  for (int r = 0; r < 16; ++r) {
    const int node = base + r;
    for (int c = lane; c < 160; c += 32) {
      float v;
      if (c < NF)        v = nf[node * NF + c];
      else if (c < 139)  v = agg[(size_t)node * HF + (c - NF)];
      else if (c == 139) v = 1.0f;
      else               v = 0.0f;
      nin[wslot][r * 160 + c] = f2bf_fast(v);
    }
  }

  // Layer 1: K=160 in 5 steps, 8 N-tiles, C starts from zero (bias in K).
  v8f acc[8];
#pragma unroll
  for (int nt = 0; nt < 8; ++nt) {
#pragma unroll
    for (int r = 0; r < 8; ++r) acc[nt][r] = 0.0f;
  }
#pragma unroll
  for (int ks = 0; ks < 5; ++ks) {
    FragAB a;
    a.q[0] = *(const uint4*)(&nin[wslot][m * 160 + ks * 32 + hi * 8]);
    a.q[1] = *(const uint4*)(&nin[wslot][m * 160 + ks * 32 + 16 + hi * 8]);
#pragma unroll
    for (int nt = 0; nt < 8; ++nt) {
      FragAB b;
      const uint4* bp = (const uint4*)(nW1T + ((nt * 16 + m) * 160 + ks * 32 + hi * 8));
      b.q[0] = bp[0]; b.q[1] = bp[2];
      acc[nt] = __builtin_amdgcn_wmma_f32_16x16x32_bf16(
          false, a.v, false, b.v, (short)0, acc[nt], false, false);
    }
  }

  // ReLU -> LDS bf16
#pragma unroll
  for (int nt = 0; nt < 8; ++nt)
#pragma unroll
    for (int r = 0; r < 8; ++r) {
      float x = __builtin_fmaxf(acc[nt][r], 0.0f);
      nh[wslot][(r + hi * 8) * HF + nt * 16 + m] = f2bf_fast(x);
    }

  // Layer 2: K=128 in 4 steps (no ReLU); bias via C splat.
  v8f acc2[8];
#pragma unroll
  for (int nt = 0; nt < 8; ++nt) {
    const float bv = nb2[nt * 16 + m];
#pragma unroll
    for (int r = 0; r < 8; ++r) acc2[nt][r] = bv;
  }
#pragma unroll
  for (int ks = 0; ks < 4; ++ks) {
    FragAB a;
    a.q[0] = *(const uint4*)(&nh[wslot][m * HF + ks * 32 + hi * 8]);
    a.q[1] = *(const uint4*)(&nh[wslot][m * HF + ks * 32 + 16 + hi * 8]);
#pragma unroll
    for (int nt = 0; nt < 8; ++nt) {
      FragAB b;
      const uint4* bp = (const uint4*)(nW2T + ((nt * 16 + m) * 128 + ks * 32 + hi * 8));
      b.q[0] = bp[0]; b.q[1] = bp[2];
      acc2[nt] = __builtin_amdgcn_wmma_f32_16x16x32_bf16(
          false, a.v, false, b.v, (short)0, acc2[nt], false, false);
    }
  }

  // Embedding head 128->4 in f32: lane holds h[M=r+8*hi][N=nt*16+m] in acc2.
  // Reduce over N within each 16-lane half via shfl_xor; fW rows as float4.
  const float4* __restrict__ fW4 = (const float4*)fW;
#pragma unroll
  for (int r = 0; r < 8; ++r) {
    float p0 = 0.f, p1 = 0.f, p2 = 0.f, p3 = 0.f;
#pragma unroll
    for (int nt = 0; nt < 8; ++nt) {
      const float hv = acc2[nt][r];
      const float4 w = fW4[nt * 16 + m];
      p0 += hv * w.x;
      p1 += hv * w.y;
      p2 += hv * w.z;
      p3 += hv * w.w;
    }
#pragma unroll
    for (int s = 8; s >= 1; s >>= 1) {
      p0 += __shfl_xor(p0, s, 32);
      p1 += __shfl_xor(p1, s, 32);
      p2 += __shfl_xor(p2, s, 32);
      p3 += __shfl_xor(p3, s, 32);
    }
    if (m == 0) {
      const int node = base + r + hi * 8;
      out[node * 4 + 0] = p0 + fb[0];
      out[node * 4 + 1] = p1 + fb[1];
      out[node * 4 + 2] = p2 + fb[2];
      out[node * 4 + 3] = p3 + fb[3];
    }
  }
}

// ---------------------------------------------------------------------------
extern "C" void kernel_launch(void* const* d_in, const int* in_sizes, int n_in,
                              void* d_out, int out_size, void* d_ws, size_t ws_size,
                              hipStream_t stream) {
  const float* nf   = (const float*)d_in[0];
  const int*   eidx = (const int*)  d_in[1];
  const float* ea   = (const float*)d_in[2];
  const float* eW1  = (const float*)d_in[3];
  const float* eb1  = (const float*)d_in[4];
  const float* eW2  = (const float*)d_in[5];
  const float* eb2  = (const float*)d_in[6];
  const float* nW1  = (const float*)d_in[7];
  const float* nb1  = (const float*)d_in[8];
  const float* nW2  = (const float*)d_in[9];
  const float* nb2  = (const float*)d_in[10];
  const float* fW   = (const float*)d_in[11];
  const float* fb   = (const float*)d_in[12];
  float* out = (float*)d_out;

  // Workspace layout: agg f32 [NN*128] | eW1T | eW2T | nW1T | nW2T (bf16)
  float* agg = (float*)d_ws;
  unsigned short* eW1T = (unsigned short*)(agg + (size_t)NN * HF);
  unsigned short* eW2T = eW1T + 128 * 32;
  unsigned short* nW1T = eW2T + 128 * 128;
  unsigned short* nW2T = nW1T + 128 * 160;

  egnn_zero_agg<<<2048, 256, 0, stream>>>(agg);
  const int wtot = 128 * 32 + 128 * 128 + 128 * 160 + 128 * 128;
  egnn_prep_weights<<<(wtot + 255) / 256, 256, 0, stream>>>(
      eW1, eb1, eW2, nW1, nb1, nW2, eW1T, eW2T, nW1T, nW2T);
  egnn_edge_kernel<<<NE / 16 / 8, 256, 0, stream>>>(
      nf, eidx, ea, eW1T, eW2T, eb2, agg);
  egnn_node_kernel<<<(NN / 16 + 3) / 4, 128, 0, stream>>>(
      nf, agg, nW1T, nW2T, nb2, fW, fb, out);
}
